// MyModel_87522843558505
// MI455X (gfx1250) — matrix operand order repaired and checked by reading, not compiled, not run on hardware
//
#include <hip/hip_runtime.h>

// CDNA5 (gfx1250, wave32) LSTM, two-phase:
//   Phase 1 (throughput): xz = x @ W + b  -> f16 workspace, WMMA GEMM over
//     M = B*T rows using the whole chip (bandwidth-bound, ~384 MB traffic).
//   Phase 2 (latency): sequential scan; per step only z = xz_t + h @ U, so
//     the 2048-step critical path carries ONE v_wmma; xz_t streams in via
//     GLOBAL_LOAD_ASYNC_TO_LDS_B128 (ASYNCcnt), double-buffered.
// Fallback: fully fused single-kernel variant if d_ws is too small.

typedef __attribute__((ext_vector_type(16))) _Float16 v16h;
typedef __attribute__((ext_vector_type(8)))  _Float16 v8h;
typedef __attribute__((ext_vector_type(8)))  float    v8f;

namespace {
constexpr int BATCH    = 256;
constexpr int TSTEPS   = 2048;
constexpr int FIN      = 128;        // input features
constexpr int H        = 32;         // LSTM units
constexpr int G        = 4 * H;      // 128 gate columns
constexpr int GP       = G + 4;      // padded f32 row (bank-conflict-free)
constexpr int BB       = 16;         // rows per block (one WMMA M-tile)
constexpr int NWAVE    = G / 16;     // 8 waves: one 16-col tile each
constexpr int NTHREADS = NWAVE * 32; // 256
}

__device__ __forceinline__ float sigmoidf_(float v) {
  return 1.0f / (1.0f + __expf(-v));
}
__device__ __forceinline__ float tanh_fast_(float v) {
  return 2.0f / (1.0f + __expf(-2.0f * v)) - 1.0f;  // v_exp/v_rcp fast path
}

// Async raw copy: 16 bytes global -> LDS per lane, tracked by ASYNCcnt.
__device__ __forceinline__ void async_copy_b128(void* lds_dst, const void* gsrc) {
  unsigned lds_off = (unsigned)(unsigned long long)lds_dst;  // low 32b = LDS byte addr
  asm volatile("global_load_async_to_lds_b128 %0, %1, off"
               :: "v"(lds_off), "v"(gsrc) : "memory");
}
__device__ __forceinline__ void async_wait0() {
  asm volatile("s_wait_asynccnt 0" ::: "memory");
}

// ---------------------------------------------------------------------------
// Phase 1: xz[m, :] = x[m, :] @ W + b   (m = b*T + t), output f16.
// Grid: (B*T)/16 blocks x 256 threads; each wave owns one 16x16 tile.
// ---------------------------------------------------------------------------
__global__ __launch_bounds__(NTHREADS)
void lstm_xw_kernel(const float* __restrict__ x,     // [B*T, F]
                    const float* __restrict__ W,     // [F, 4H]
                    const float* __restrict__ bias,  // [4H]
                    _Float16* __restrict__ xz)       // [B*T, 4H] f16
{
  __shared__ __align__(16) _Float16 s_a[BB][FIN];   // x rows as f16
  __shared__ __align__(16) float    s_o[BB][GP];    // f32 result staging

  const int tid    = threadIdx.x;
  const int wave   = tid >> 5;
  const int lane   = tid & 31;
  const int nloc   = lane & 15;
  const int hiHalf = (lane >= 16) ? 1 : 0;
  const size_t m0  = (size_t)blockIdx.x * BB;

  // B-tile layout: lanes 0-15 K=0..15, lanes 16-31 K=16..31, b[j]=K=kb+j
  const int ncol = wave * 16 + nloc;
  const int kb   = hiHalf ? 16 : 0;
  v16h Wt[4];
#pragma unroll
  for (int kc = 0; kc < 4; ++kc) {
    v16h w;
#pragma unroll
    for (int j = 0; j < 16; ++j)
      w[j] = (_Float16)W[(kc * 32 + kb + j) * G + ncol];
    Wt[kc] = w;
  }
  const float bn = bias[ncol];

  // stage 16 rows of x (f32 -> f16): 512 float4, 2 per thread
  {
    const int r0 = tid >> 5, c0 = tid & 31;
    const int r1 = r0 + 8;
    float4 v0 = ((const float4*)(x + (m0 + r0) * FIN))[c0];
    float4 v1 = ((const float4*)(x + (m0 + r1) * FIN))[c0];
    _Float16* d0 = &s_a[r0][c0 * 4];
    d0[0] = (_Float16)v0.x; d0[1] = (_Float16)v0.y;
    d0[2] = (_Float16)v0.z; d0[3] = (_Float16)v0.w;
    _Float16* d1 = &s_a[r1][c0 * 4];
    d1[0] = (_Float16)v1.x; d1[1] = (_Float16)v1.y;
    d1[2] = (_Float16)v1.z; d1[3] = (_Float16)v1.w;
  }
  __syncthreads();

  // A layout: lane row m=lane%16; lanes 0-15 = K{0-15},
  // lanes 16-31 = K{16-31} (per 32-K chunk)
  const int am = nloc;
  const int ab = hiHalf;
  const int mr = hiHalf ? 8 : 0;

  v16h A[4];
  const v8h* xrow = (const v8h*)&s_a[am][0];
#pragma unroll
  for (int kc = 0; kc < 4; ++kc) {
    v8h lo = xrow[kc * 4 + 2 * ab];
    v8h hi = xrow[kc * 4 + 2 * ab + 1];
#pragma unroll
    for (int j = 0; j < 8; ++j) { A[kc][j] = lo[j]; A[kc][j + 8] = hi[j]; }
  }

  // two independent accumulator chains (shorter dependency)
  v8f acc0, acc1;
#pragma unroll
  for (int v = 0; v < 8; ++v) { acc0[v] = bn; acc1[v] = 0.0f; }
  acc0 = __builtin_amdgcn_wmma_f32_16x16x32_f16(false, A[0], false, Wt[0],
                                                (short)0, acc0, false, false);
  acc1 = __builtin_amdgcn_wmma_f32_16x16x32_f16(false, A[1], false, Wt[1],
                                                (short)0, acc1, false, false);
  acc0 = __builtin_amdgcn_wmma_f32_16x16x32_f16(false, A[2], false, Wt[2],
                                                (short)0, acc0, false, false);
  acc1 = __builtin_amdgcn_wmma_f32_16x16x32_f16(false, A[3], false, Wt[3],
                                                (short)0, acc1, false, false);
#pragma unroll
  for (int v = 0; v < 8; ++v) s_o[mr + v][ncol] = acc0[v] + acc1[v];
  __syncthreads();

  // pack to f16 and store coalesced: 16 threads/row x 8 cols
  {
    const int r  = tid >> 4;
    const int cs = (tid & 15) * 8;
    v8h o;
#pragma unroll
    for (int j = 0; j < 8; ++j) o[j] = (_Float16)s_o[r][cs + j];
    *(v8h*)(xz + (m0 + r) * G + cs) = o;
  }
}

// ---------------------------------------------------------------------------
// Phase 2: sequential scan.  z_t = xz_t + h @ U ; gates ; h,c in LDS.
// xz_t staged by async global->LDS copies (ASYNCcnt), double-buffered.
// Grid: 16 blocks (16 batch rows each) x 256 threads.
// ---------------------------------------------------------------------------
__global__ __launch_bounds__(NTHREADS)
void lstm_scan_kernel(const _Float16* __restrict__ xz,  // [B*T, 4H] f16
                      const float* __restrict__ U,      // [H, 4H]
                      float* __restrict__ out)          // [B, H]
{
  __shared__ __align__(16) _Float16 s_xz[2][BB][G];   // double-buffered xz_t
  __shared__ __align__(16) float    s_z[BB][GP];      // h@U partial (f32)
  __shared__ __align__(16) float    s_c[BB][H];
  __shared__ __align__(16) _Float16 s_h[BB][H];

  const int tid    = threadIdx.x;
  const int wave   = tid >> 5;
  const int lane   = tid & 31;
  const int nloc   = lane & 15;
  const int hiHalf = (lane >= 16) ? 1 : 0;
  const int b0     = blockIdx.x * BB;

  const int ncol = wave * 16 + nloc;
  const int kb   = hiHalf ? 16 : 0;
  v16h Ut;
#pragma unroll
  for (int j = 0; j < 16; ++j)
    Ut[j] = (_Float16)U[(kb + j) * G + ncol];

  // xz staging: 16 threads/row, one b128 (16B) each
  const int rs = tid >> 4;
  const int cs = (tid & 15) * 8;
  const _Float16* xzrow = xz + ((size_t)(b0 + rs) * TSTEPS) * G + cs;
  // gate work: rows rg, rg+8, column cg
  const int rg = tid >> 5, cg = tid & 31;

  s_c[rg][cg]     = 0.0f;
  s_c[rg + 8][cg] = 0.0f;
  s_h[rg][cg]     = (_Float16)0.0f;
  s_h[rg + 8][cg] = (_Float16)0.0f;
  async_copy_b128(&s_xz[0][rs][cs], xzrow);
  async_wait0();
  __syncthreads();

  const int am = nloc;
  const int ab = hiHalf;
  const int mr = hiHalf ? 8 : 0;

  for (int t = 0; t < TSTEPS; ++t) {
    const int buf  = t & 1;
    const int nbuf = buf ^ 1;
    const int tp   = (t + 1 < TSTEPS) ? (t + 1) : t;

    // async prefetch of next xz tile straight into LDS (off critical path)
    async_copy_b128(&s_xz[nbuf][rs][cs], xzrow + (size_t)tp * G);

    // ---- critical path: ONE wmma (h @ U) ----
    v16h Ah;
    {
      const v8h* hrow = (const v8h*)&s_h[am][0];
      v8h lo = hrow[2 * ab];
      v8h hi = hrow[2 * ab + 1];
#pragma unroll
      for (int j = 0; j < 8; ++j) { Ah[j] = lo[j]; Ah[j + 8] = hi[j]; }
    }
    v8f acc = {};
    acc = __builtin_amdgcn_wmma_f32_16x16x32_f16(false, Ah, false, Ut,
                                                 (short)0, acc, false, false);
#pragma unroll
    for (int v = 0; v < 8; ++v) s_z[mr + v][ncol] = acc[v];

    async_wait0();        // prefetch landed in LDS (needed after next barrier)
    __syncthreads();

    // ---- gates: z = xz_t + hU ; i,f,g,o column blocks of 32 ----
#pragma unroll
    for (int half = 0; half < 2; ++half) {
      const int r = rg + half * 8;
      const float zi = (float)s_xz[buf][r][cg]         + s_z[r][cg];
      const float zf = (float)s_xz[buf][r][cg + H]     + s_z[r][cg + H];
      const float zg = (float)s_xz[buf][r][cg + 2 * H] + s_z[r][cg + 2 * H];
      const float zo = (float)s_xz[buf][r][cg + 3 * H] + s_z[r][cg + 3 * H];
      const float cn = sigmoidf_(zf) * s_c[r][cg] +
                       sigmoidf_(zi) * tanh_fast_(zg);
      s_c[r][cg] = cn;
      const float hn = sigmoidf_(zo) * tanh_fast_(cn);
      s_h[r][cg] = (_Float16)hn;
      if (t == TSTEPS - 1) out[(size_t)(b0 + r) * H + cg] = hn;
    }
    __syncthreads();
  }
}

// ---------------------------------------------------------------------------
// Fallback: fully fused kernel (used only if workspace is too small).
// ---------------------------------------------------------------------------
__global__ __launch_bounds__(NTHREADS)
void lstm_fused_kernel(const float* __restrict__ x, const float* __restrict__ W,
                       const float* __restrict__ U, const float* __restrict__ bias,
                       float* __restrict__ out)
{
  __shared__ __align__(16) _Float16 s_x[2][BB][FIN];
  __shared__ __align__(16) float    s_z[BB][GP];
  __shared__ __align__(16) float    s_c[BB][H];
  __shared__ __align__(16) _Float16 s_h[BB][H];

  const int tid    = threadIdx.x;
  const int wave   = tid >> 5;
  const int lane   = tid & 31;
  const int nloc   = lane & 15;
  const int hiHalf = (lane >= 16) ? 1 : 0;
  const int b0     = blockIdx.x * BB;

  const int ncol = wave * 16 + nloc;
  const int kb   = hiHalf ? 16 : 0;
  v16h Wt[4];
#pragma unroll
  for (int kc = 0; kc < 4; ++kc) {
    v16h w;
#pragma unroll
    for (int j = 0; j < 16; ++j)
      w[j] = (_Float16)W[(kc * 32 + kb + j) * G + ncol];
    Wt[kc] = w;
  }
  v16h Ut;
#pragma unroll
  for (int j = 0; j < 16; ++j)
    Ut[j] = (_Float16)U[(kb + j) * G + ncol];
  const float bn = bias[ncol];

  const int r0 = tid >> 5, c0 = tid & 31;
  const int r1 = r0 + 8;
  const float* xp0 = x + (size_t)(b0 + r0) * TSTEPS * FIN;
  const float* xp1 = x + (size_t)(b0 + r1) * TSTEPS * FIN;
  const int rg = tid >> 5, cg = tid & 31;

  s_c[rg][cg]     = 0.0f;
  s_c[rg + 8][cg] = 0.0f;
  s_h[rg][cg]     = (_Float16)0.0f;
  s_h[rg + 8][cg] = (_Float16)0.0f;
  {
    float4 v0 = ((const float4*)xp0)[c0];
    float4 v1 = ((const float4*)xp1)[c0];
    _Float16* d0 = &s_x[0][r0][c0 * 4];
    d0[0] = (_Float16)v0.x; d0[1] = (_Float16)v0.y;
    d0[2] = (_Float16)v0.z; d0[3] = (_Float16)v0.w;
    _Float16* d1 = &s_x[0][r1][c0 * 4];
    d1[0] = (_Float16)v1.x; d1[1] = (_Float16)v1.y;
    d1[2] = (_Float16)v1.z; d1[3] = (_Float16)v1.w;
  }
  __syncthreads();

  const int am = nloc;
  const int ab = hiHalf;
  const int mr = hiHalf ? 8 : 0;

  for (int t = 0; t < TSTEPS; ++t) {
    const int buf  = t & 1;
    const int nbuf = buf ^ 1;
    const int tp   = (t + 1 < TSTEPS) ? (t + 1) : t;

    float4 pf0 = ((const float4*)(xp0 + (size_t)tp * FIN))[c0];
    float4 pf1 = ((const float4*)(xp1 + (size_t)tp * FIN))[c0];

    v16h A[5];
    const v8h* xrow = (const v8h*)&s_x[buf][am][0];
#pragma unroll
    for (int kc = 0; kc < 4; ++kc) {
      v8h lo = xrow[kc * 4 + 2 * ab];
      v8h hi = xrow[kc * 4 + 2 * ab + 1];
#pragma unroll
      for (int j = 0; j < 8; ++j) { A[kc][j] = lo[j]; A[kc][j + 8] = hi[j]; }
    }
    {
      const v8h* hrow = (const v8h*)&s_h[am][0];
      v8h lo = hrow[2 * ab];
      v8h hi = hrow[2 * ab + 1];
#pragma unroll
      for (int j = 0; j < 8; ++j) { A[4][j] = lo[j]; A[4][j + 8] = hi[j]; }
    }

    v8f acc0, acc1;
#pragma unroll
    for (int v = 0; v < 8; ++v) { acc0[v] = bn; acc1[v] = 0.0f; }
    acc0 = __builtin_amdgcn_wmma_f32_16x16x32_f16(false, A[0], false, Wt[0],
                                                  (short)0, acc0, false, false);
    acc1 = __builtin_amdgcn_wmma_f32_16x16x32_f16(false, A[1], false, Wt[1],
                                                  (short)0, acc1, false, false);
    acc0 = __builtin_amdgcn_wmma_f32_16x16x32_f16(false, A[2], false, Wt[2],
                                                  (short)0, acc0, false, false);
    acc1 = __builtin_amdgcn_wmma_f32_16x16x32_f16(false, A[3], false, Wt[3],
                                                  (short)0, acc1, false, false);
    acc0 = __builtin_amdgcn_wmma_f32_16x16x32_f16(false, A[4], false, Ut,
                                                  (short)0, acc0, false, false);
#pragma unroll
    for (int v = 0; v < 8; ++v) s_z[mr + v][ncol] = acc0[v] + acc1[v];

    {
      _Float16* d0 = &s_x[nbuf][r0][c0 * 4];
      d0[0] = (_Float16)pf0.x; d0[1] = (_Float16)pf0.y;
      d0[2] = (_Float16)pf0.z; d0[3] = (_Float16)pf0.w;
      _Float16* d1 = &s_x[nbuf][r1][c0 * 4];
      d1[0] = (_Float16)pf1.x; d1[1] = (_Float16)pf1.y;
      d1[2] = (_Float16)pf1.z; d1[3] = (_Float16)pf1.w;
    }
    __syncthreads();

#pragma unroll
    for (int half = 0; half < 2; ++half) {
      const int r = rg + half * 8;
      const float zi = s_z[r][cg];
      const float zf = s_z[r][cg + H];
      const float zg = s_z[r][cg + 2 * H];
      const float zo = s_z[r][cg + 3 * H];
      const float cn = sigmoidf_(zf) * s_c[r][cg] +
                       sigmoidf_(zi) * tanh_fast_(zg);
      s_c[r][cg] = cn;
      const float hn = sigmoidf_(zo) * tanh_fast_(cn);
      s_h[r][cg] = (_Float16)hn;
      if (t == TSTEPS - 1) out[(size_t)(b0 + r) * H + cg] = hn;
    }
    __syncthreads();
  }
}

extern "C" void kernel_launch(void* const* d_in, const int* in_sizes, int n_in,
                              void* d_out, int out_size, void* d_ws, size_t ws_size,
                              hipStream_t stream) {
  (void)in_sizes; (void)n_in; (void)out_size;
  const float* x  = (const float*)d_in[0];
  const float* W  = (const float*)d_in[1];
  const float* U  = (const float*)d_in[2];
  const float* b  = (const float*)d_in[3];
  float* out = (float*)d_out;

  const size_t need = (size_t)BATCH * TSTEPS * G * 2 /* f16 */;
  if (ws_size >= need && d_ws != nullptr) {
    _Float16* xz = (_Float16*)d_ws;
    lstm_xw_kernel<<<dim3((BATCH * TSTEPS) / BB), dim3(NTHREADS), 0, stream>>>(
        x, W, b, xz);
    lstm_scan_kernel<<<dim3(BATCH / BB), dim3(NTHREADS), 0, stream>>>(
        xz, U, out);
  } else {
    lstm_fused_kernel<<<dim3(BATCH / BB), dim3(NTHREADS), 0, stream>>>(
        x, W, U, b, out);
  }
}
